// CapacitiveMHA_2181843387016
// MI455X (gfx1250) — matrix operand order repaired and checked by reading, not compiled, not run on hardware
//
#include <hip/hip_runtime.h>
#include <hip/hip_bf16.h>
#include <math.h>

// ---------------------------------------------------------------------------
// CapacitiveMHA for MI455X (gfx1250, wave32, WMMA).
// Pipeline: weight->bf16, router, LDS bitonic top-k, KV GEMM(+RoPE K, V^T),
// gathered Q GEMM(+RoPE, 1/sqrt(d)), flash-attention with cooperative
// double-buffered GLOBAL_LOAD_ASYNC_TO_LDS_B128 K/V staging (ASYNCcnt) +
// WMMA QK^T + online softmax + LDS P relayout + WMMA PV, out-proj + scatter.
// All matrix math: v_wmma_f32_16x16x32_bf16 (f32 accumulate).
// ---------------------------------------------------------------------------

typedef __bf16 bf16_t;
typedef bf16_t v16bf __attribute__((ext_vector_type(16)));
typedef float  v8f   __attribute__((ext_vector_type(8)));

#define BATCH    4
#define SEQ      4096
#define DMODEL   1024
#define HEADS    16
#define CAP      512
#define DHEAD    64
#define LOG_BASE 9.210340372f /* ln(10000) */

// ---- WMMA fragment index maps (ISA 7.12.2, wave32) -------------------------
__device__ __forceinline__ int kb_a(int j, int half) {
  return (j < 4) ? (2 * j + 8 * half) : (16 + 2 * (j - 4) + 8 * half);
}
__device__ __forceinline__ int kb_b(int j, int half) {
  return (j < 4) ? (2 * j + 8 * half) : (16 + 2 * (j - 4) + 8 * half);
}

__device__ __forceinline__ v8f wmma_bf16(v16bf a, v16bf b, v8f c) {
  return __builtin_amdgcn_wmma_f32_16x16x32_bf16(false, a, false, b, (short)0, c,
                                                 false, false);
}

// A-fragment (16xK tile, rows in lanes) from fp32 row (convert) or bf16 row.
__device__ __forceinline__ v16bf load_a_f32(const float* row, int k0, int half) {
  v16bf a;
#pragma unroll
  for (int j = 0; j < 8; ++j) {
    int kb = k0 + kb_a(j, half);
    a[2 * j]     = (bf16_t)row[kb];
    a[2 * j + 1] = (bf16_t)row[kb + 1];
  }
  return a;
}
__device__ __forceinline__ v16bf load_a_bf16(const bf16_t* row, int k0, int half) {
  v16bf a;
#pragma unroll
  for (int j = 0; j < 8; ++j) {
    int kb = k0 + kb_a(j, half);
    a[2 * j]     = row[kb];
    a[2 * j + 1] = row[kb + 1];
  }
  return a;
}
// B-fragment for B[k][n] = M[n0+n][k0+k] with M row-major, row stride ld
// (covers W^T for projections, K^T tiles, and V^T tiles).
__device__ __forceinline__ v16bf load_b(const bf16_t* M, int ld, int n0, int k0,
                                        int lane) {
  int n = lane & 15, half = lane >> 4;
  const bf16_t* p = M + (size_t)(n0 + n) * ld + k0;
  v16bf b;
#pragma unroll
  for (int j = 0; j < 8; ++j) {
    int kb = kb_b(j, half);
    b[2 * j]     = p[kb];
    b[2 * j + 1] = p[kb + 1];
  }
  return b;
}
__device__ __forceinline__ v8f v8f_zero() {
  v8f z;
#pragma unroll
  for (int r = 0; r < 8; ++r) z[r] = 0.0f;
  return z;
}
__device__ __forceinline__ float rope_factor(float pos, int t) {
  int i = t & 31;
  float freq = __expf(-(float)i * (LOG_BASE / 31.0f));
  float ang = pos * freq;
  return (t < 32) ? __sinf(ang) : __cosf(ang);
}

// Async 16-byte global->LDS copy (ASYNCcnt-tracked, ISA §10 / §15.18.3).
// Low 32 bits of a generic pointer to LDS are the LDS byte address.
__device__ __forceinline__ void async_copy16(const bf16_t* gptr, bf16_t* lptr) {
  unsigned int lds = (unsigned int)(size_t)lptr;
  asm volatile("global_load_async_to_lds_b128 %0, %1, off"
               :
               : "v"(lds), "v"(gptr)
               : "memory");
}

// ---- small utility kernels -------------------------------------------------
__global__ void zero_f32_kernel(float* p, size_t n) {
  size_t i = (size_t)blockIdx.x * blockDim.x + threadIdx.x;
  size_t st = (size_t)gridDim.x * blockDim.x;
  for (; i < n; i += st) p[i] = 0.0f;
}
__global__ void f32_to_bf16_kernel(const float* s, bf16_t* d, int n) {
  int i = blockIdx.x * blockDim.x + threadIdx.x;
  int st = gridDim.x * blockDim.x;
  for (; i < n; i += st) d[i] = (bf16_t)s[i];
}

// rw[b,s] = dot(query[b,s,:], router_w)
__global__ __launch_bounds__(256) void router_kernel(const float* __restrict__ q,
                                                     const float* __restrict__ w,
                                                     float* __restrict__ rw) {
  int t = blockIdx.x * 256 + threadIdx.x;  // 16384 rows
  const float4* qp = (const float4*)(q + (size_t)t * DMODEL);
  const float4* wp = (const float4*)w;
  float s = 0.0f;
#pragma unroll 4
  for (int i = 0; i < DMODEL / 4; ++i) {
    float4 a = qp[i], b = wp[i];
    s = fmaf(a.x, b.x, fmaf(a.y, b.y, fmaf(a.z, b.z, fmaf(a.w, b.w, s))));
  }
  rw[t] = s;
}

// Full LDS bitonic sort (desc by value, asc index tiebreak), keep top 512.
__global__ __launch_bounds__(1024) void topk_kernel(const float* __restrict__ rw,
                                                    float* __restrict__ top_w,
                                                    int* __restrict__ top_idx) {
  __shared__ float sk[SEQ];
  __shared__ int si[SEQ];
  int b = blockIdx.x, tid = threadIdx.x;
  for (int i = tid; i < SEQ; i += 1024) {
    sk[i] = rw[b * SEQ + i];
    si[i] = i;
  }
  for (int k = 2; k <= SEQ; k <<= 1) {
    for (int j = k >> 1; j > 0; j >>= 1) {
      __syncthreads();
      for (int base = 0; base < SEQ; base += 1024) {
        int i = base + tid;
        int ixj = i ^ j;
        if (ixj > i) {
          float a = sk[i], c = sk[ixj];
          int ia = si[i], ic = si[ixj];
          bool firstWins = (a > c) || (a == c && ia < ic);
          bool descBlock = ((i & k) == 0);
          if (descBlock ? !firstWins : firstWins) {
            sk[i] = c; sk[ixj] = a; si[i] = ic; si[ixj] = ia;
          }
        }
      }
    }
  }
  __syncthreads();
  if (tid < CAP) {
    top_w[b * CAP + tid] = sk[tid];
    top_idx[b * CAP + tid] = si[tid];
  }
}

// KV = value @ kv_w^T ; K gets RoPE, stored [b,h,s,d]; V stored [b,h,d,s].
__global__ __launch_bounds__(256) void kv_gemm_kernel(const float* __restrict__ value,
                                                      const bf16_t* __restrict__ kvw,
                                                      bf16_t* __restrict__ Kb,
                                                      bf16_t* __restrict__ Vb) {
  int wv = blockIdx.x * 8 + (threadIdx.x >> 5);  // 32768 waves
  int lane = threadIdx.x & 31;
  int mTile = wv >> 5;   // 1024 M-tiles of 16
  int nT = wv & 31;      // 32 N-tiles of 64
  int row0 = mTile * 16, n0 = nT * 64;
  int m = lane & 15, half = lane >> 4, nlane = m;
  const float* arow = value + (size_t)(row0 + m) * DMODEL;
  v8f acc[4];
#pragma unroll
  for (int t = 0; t < 4; ++t) acc[t] = v8f_zero();
  for (int k0 = 0; k0 < DMODEL; k0 += 32) {
    __builtin_prefetch(arow + ((k0 + 64) & (DMODEL - 1)), 0, 0);
    v16bf a = load_a_f32(arow, k0, half);
#pragma unroll
    for (int t = 0; t < 4; ++t) {
      v16bf b = load_b(kvw, DMODEL, n0 + 16 * t, k0, lane);
      acc[t] = wmma_bf16(a, b, acc[t]);
    }
  }
#pragma unroll
  for (int r = 0; r < 8; ++r) {
    int grow = row0 + r + 8 * half;
    int bb = grow >> 12, s = grow & (SEQ - 1);
#pragma unroll
    for (int t = 0; t < 4; ++t) {
      int e = n0 + 16 * t + nlane;
      float v = acc[t][r];
      if (e < DMODEL) {  // K half (+RoPE)
        int h = e >> 6, tt = e & 63;
        float f = rope_factor((float)s, tt);
        Kb[(((size_t)bb * HEADS + h) * SEQ + s) * DHEAD + tt] = (bf16_t)(v * f);
      } else {           // V half, transposed store
        int e2 = e - DMODEL;
        int h = e2 >> 6, tt = e2 & 63;
        Vb[(((size_t)bb * HEADS + h) * DHEAD + tt) * SEQ + s] = (bf16_t)v;
      }
    }
  }
}

// Q = gather(query, top_idx) @ q_w^T ; RoPE(pos=idx) * 1/sqrt(d); [b,h,c,d].
__global__ __launch_bounds__(256) void q_gemm_kernel(const float* __restrict__ query,
                                                     const bf16_t* __restrict__ qw,
                                                     const int* __restrict__ top_idx,
                                                     bf16_t* __restrict__ Qb) {
  int wv = blockIdx.x * 8 + (threadIdx.x >> 5);  // 2048 waves
  int lane = threadIdx.x & 31;
  int mTile = wv >> 4;  // 128 M-tiles
  int nT = wv & 15;     // 16 N-tiles of 64
  int row0 = mTile * 16, n0 = nT * 64;
  int m = lane & 15, half = lane >> 4, nlane = m;
  int growA = row0 + m;
  int bbA = growA >> 9;
  const float* arow = query + ((size_t)bbA * SEQ + top_idx[growA]) * DMODEL;
  v8f acc[4];
#pragma unroll
  for (int t = 0; t < 4; ++t) acc[t] = v8f_zero();
  for (int k0 = 0; k0 < DMODEL; k0 += 32) {
    v16bf a = load_a_f32(arow, k0, half);
#pragma unroll
    for (int t = 0; t < 4; ++t) {
      v16bf b = load_b(qw, DMODEL, n0 + 16 * t, k0, lane);
      acc[t] = wmma_bf16(a, b, acc[t]);
    }
  }
#pragma unroll
  for (int r = 0; r < 8; ++r) {
    int grow = row0 + r + 8 * half;
    int bb = grow >> 9, c = grow & (CAP - 1);
    float pos = (float)top_idx[grow];
#pragma unroll
    for (int t = 0; t < 4; ++t) {
      int e = n0 + 16 * t + nlane;
      int h = e >> 6, tt = e & 63;
      float v = acc[t][r] * rope_factor(pos, tt) * 0.125f;  // 1/sqrt(64) folded
      Qb[(((size_t)bb * HEADS + h) * CAP + c) * DHEAD + tt] = (bf16_t)v;
    }
  }
}

// Flash attention. Block = 8 waves, all on ONE (b,h): 128 queries/block.
// K/V tiles (32 keys) are staged cooperatively into LDS with double-buffered
// async global->LDS copies; fragments are then read from LDS (ds ops).
__global__ __launch_bounds__(256) void attn_kernel(const bf16_t* __restrict__ Qb,
                                                   const bf16_t* __restrict__ Kb,
                                                   const bf16_t* __restrict__ Vb,
                                                   bf16_t* __restrict__ attb) {
  __shared__ bf16_t Kt[2][32 * DHEAD];   // 2 x 4KB
  __shared__ bf16_t Vt[2][DHEAD * 32];   // 2 x 4KB
  __shared__ bf16_t psh_all[8][16 * 32]; // 8KB, per-wave P relayout buffer
  int tid = threadIdx.x;
  int wid = tid >> 5, lane = tid & 31;
  int bh = blockIdx.x >> 2;              // 64 (b,h) pairs
  int qt = (blockIdx.x & 3) * 8 + wid;   // 32 q-tiles per (b,h)
  bf16_t* psh = psh_all[wid];
  int m = lane & 15, half = lane >> 4, nlane = m;
  const bf16_t* Qp = Qb + ((size_t)bh * CAP + qt * 16) * DHEAD;
  const bf16_t* Kp = Kb + (size_t)bh * SEQ * DHEAD;
  const bf16_t* Vp = Vb + (size_t)bh * DHEAD * SEQ;

  const bf16_t* qrow = Qp + (size_t)m * DHEAD;
  v16bf qa0 = load_a_bf16(qrow, 0, half);
  v16bf qa1 = load_a_bf16(qrow, 32, half);

  v8f o[4];
#pragma unroll
  for (int t = 0; t < 4; ++t) o[t] = v8f_zero();
  float mrow[8], lrow[8];
#pragma unroll
  for (int r = 0; r < 8; ++r) { mrow[r] = -1.0e30f; lrow[r] = 0.0f; }

  // per-thread async-copy assignments (256 threads, 16B each):
  int kr = tid >> 3, kc = tid & 7;  // K tile: 32 rows x 64 bf16 (8 chunks/row)
  int vr = tid >> 2, vc = tid & 3;  // V tile: 64 rows x 32 bf16 (4 chunks/row)

  // prime buffer 0
  async_copy16(Kp + (size_t)kr * DHEAD + kc * 8, &Kt[0][kr * 64 + kc * 8]);
  async_copy16(Vp + (size_t)vr * SEQ + vc * 8, &Vt[0][vr * 32 + vc * 8]);

  for (int it = 0; it < SEQ / 32; ++it) {
    int cur = it & 1;
    if (it + 1 < SEQ / 32) {
      int s1 = (it + 1) * 32;
      async_copy16(Kp + (size_t)(s1 + kr) * DHEAD + kc * 8,
                   &Kt[cur ^ 1][kr * 64 + kc * 8]);
      async_copy16(Vp + (size_t)vr * SEQ + s1 + vc * 8,
                   &Vt[cur ^ 1][vr * 32 + vc * 8]);
      asm volatile("s_wait_asynccnt 0x2" ::: "memory");  // only current buffer
    } else {
      asm volatile("s_wait_asynccnt 0x0" ::: "memory");
    }
    __syncthreads();
    const bf16_t* kt = Kt[cur];
    const bf16_t* vt = Vt[cur];

    v8f sc0 = v8f_zero(), sc1 = v8f_zero();
    v16bf kf;
    kf = load_b(kt, DHEAD, 0, 0, lane);   sc0 = wmma_bf16(qa0, kf, sc0);
    kf = load_b(kt, DHEAD, 0, 32, lane);  sc0 = wmma_bf16(qa1, kf, sc0);
    kf = load_b(kt, DHEAD, 16, 0, lane);  sc1 = wmma_bf16(qa0, kf, sc1);
    kf = load_b(kt, DHEAD, 16, 32, lane); sc1 = wmma_bf16(qa1, kf, sc1);

    // online softmax; row m=r+8*half lives across the 16 lanes of one half
#pragma unroll
    for (int r = 0; r < 8; ++r) {
      float v0 = sc0[r], v1 = sc1[r];
      float mx = fmaxf(v0, v1);
      mx = fmaxf(mx, __shfl_xor(mx, 1));
      mx = fmaxf(mx, __shfl_xor(mx, 2));
      mx = fmaxf(mx, __shfl_xor(mx, 4));
      mx = fmaxf(mx, __shfl_xor(mx, 8));
      float mnew = fmaxf(mrow[r], mx);
      float p0 = __expf(v0 - mnew), p1 = __expf(v1 - mnew);
      float ps = p0 + p1;
      ps += __shfl_xor(ps, 1);
      ps += __shfl_xor(ps, 2);
      ps += __shfl_xor(ps, 4);
      ps += __shfl_xor(ps, 8);
      float sc = __expf(mrow[r] - mnew);
      lrow[r] = lrow[r] * sc + ps;
      mrow[r] = mnew;
      o[0][r] *= sc; o[1][r] *= sc; o[2][r] *= sc; o[3][r] *= sc;
      sc0[r] = p0; sc1[r] = p1;
    }
    // C-layout P -> LDS -> A-fragment relayout (wave-private region)
#pragma unroll
    for (int r = 0; r < 8; ++r) {
      int mm = r + 8 * half;
      psh[mm * 32 + nlane] = (bf16_t)sc0[r];
      psh[mm * 32 + 16 + nlane] = (bf16_t)sc1[r];
    }
    asm volatile("s_wait_dscnt 0" ::: "memory");
    v16bf pa;
#pragma unroll
    for (int j = 0; j < 8; ++j) {
      int kb = kb_a(j, half);
      pa[2 * j]     = psh[m * 32 + kb];
      pa[2 * j + 1] = psh[m * 32 + kb + 1];
    }
#pragma unroll
    for (int t = 0; t < 4; ++t) {
      v16bf vf = load_b(vt, 32, t * 16, 0, lane);
      o[t] = wmma_bf16(pa, vf, o[t]);
    }
    __syncthreads();  // all waves done with this buffer before it is refilled
  }
  int b = bh >> 4, h = bh & 15;
#pragma unroll
  for (int r = 0; r < 8; ++r) {
    int mm = r + 8 * half;
    int c = qt * 16 + mm;
    float inv = 1.0f / lrow[r];
#pragma unroll
    for (int t = 0; t < 4; ++t) {
      int tt = t * 16 + nlane;
      attb[((size_t)b * CAP + c) * DMODEL + h * DHEAD + tt] = (bf16_t)(o[t][r] * inv);
    }
  }
}

// out = (att @ out_w^T) * top_w, scattered to d_out rows top_idx.
__global__ __launch_bounds__(256) void outproj_kernel(const bf16_t* __restrict__ attb,
                                                      const bf16_t* __restrict__ ow,
                                                      const float* __restrict__ top_w,
                                                      const int* __restrict__ top_idx,
                                                      float* __restrict__ out) {
  int wv = blockIdx.x * 8 + (threadIdx.x >> 5);  // 2048 waves
  int lane = threadIdx.x & 31;
  int mTile = wv >> 4, nT = wv & 15;
  int row0 = mTile * 16, n0 = nT * 64;
  int m = lane & 15, half = lane >> 4, nlane = m;
  const bf16_t* arow = attb + (size_t)(row0 + m) * DMODEL;
  v8f acc[4];
#pragma unroll
  for (int t = 0; t < 4; ++t) acc[t] = v8f_zero();
  for (int k0 = 0; k0 < DMODEL; k0 += 32) {
    v16bf a = load_a_bf16(arow, k0, half);
#pragma unroll
    for (int t = 0; t < 4; ++t) {
      v16bf b = load_b(ow, DMODEL, n0 + 16 * t, k0, lane);
      acc[t] = wmma_bf16(a, b, acc[t]);
    }
  }
#pragma unroll
  for (int r = 0; r < 8; ++r) {
    int grow = row0 + r + 8 * half;
    int bb = grow >> 9;
    int idx = top_idx[grow];
    float w = top_w[grow];
#pragma unroll
    for (int t = 0; t < 4; ++t) {
      int col = n0 + 16 * t + nlane;
      out[(((size_t)bb * SEQ + idx) << 10) + col] = acc[t][r] * w;
    }
  }
}

// ---------------------------------------------------------------------------
extern "C" void kernel_launch(void* const* d_in, const int* in_sizes, int n_in,
                              void* d_out, int out_size, void* d_ws, size_t ws_size,
                              hipStream_t stream) {
  (void)in_sizes; (void)n_in; (void)out_size; (void)ws_size;
  const float* query    = (const float*)d_in[0];
  const float* value    = (const float*)d_in[1];
  const float* router_w = (const float*)d_in[2];
  const float* q_w      = (const float*)d_in[3];
  const float* kv_w     = (const float*)d_in[4];
  const float* out_w    = (const float*)d_in[5];
  float* out = (float*)d_out;

  char* ws = (char*)d_ws;
  size_t off = 0;
  auto carve = [&](size_t bytes) -> void* {
    void* p = ws + off;
    off = (off + bytes + 255) & ~(size_t)255;
    return p;
  };
  float*  rw      = (float*)carve((size_t)BATCH * SEQ * 4);
  float*  top_w   = (float*)carve((size_t)BATCH * CAP * 4);
  int*    top_idx = (int*)carve((size_t)BATCH * CAP * 4);
  bf16_t* kvw_bf  = (bf16_t*)carve((size_t)2 * DMODEL * DMODEL * 2);
  bf16_t* qw_bf   = (bf16_t*)carve((size_t)DMODEL * DMODEL * 2);
  bf16_t* ow_bf   = (bf16_t*)carve((size_t)DMODEL * DMODEL * 2);
  bf16_t* Kb      = (bf16_t*)carve((size_t)BATCH * HEADS * SEQ * DHEAD * 2);
  bf16_t* Vb      = (bf16_t*)carve((size_t)BATCH * HEADS * SEQ * DHEAD * 2);
  bf16_t* Qb      = (bf16_t*)carve((size_t)BATCH * HEADS * CAP * DHEAD * 2);
  bf16_t* attb    = (bf16_t*)carve((size_t)BATCH * CAP * DMODEL * 2);

  f32_to_bf16_kernel<<<2048, 256, 0, stream>>>(kv_w, kvw_bf, 2 * DMODEL * DMODEL);
  f32_to_bf16_kernel<<<1024, 256, 0, stream>>>(q_w, qw_bf, DMODEL * DMODEL);
  f32_to_bf16_kernel<<<1024, 256, 0, stream>>>(out_w, ow_bf, DMODEL * DMODEL);
  router_kernel<<<(BATCH * SEQ) / 256, 256, 0, stream>>>(query, router_w, rw);
  topk_kernel<<<BATCH, 1024, 0, stream>>>(rw, top_w, top_idx);
  kv_gemm_kernel<<<4096, 256, 0, stream>>>(value, kvw_bf, Kb, Vb);
  q_gemm_kernel<<<256, 256, 0, stream>>>(query, qw_bf, top_idx, Qb);
  attn_kernel<<<256, 256, 0, stream>>>(Qb, Kb, Vb, attb);
  zero_f32_kernel<<<2048, 256, 0, stream>>>(out, (size_t)BATCH * SEQ * DMODEL);
  outproj_kernel<<<256, 256, 0, stream>>>(attb, ow_bf, top_w, top_idx, out);
}
